// SparseGraphLinkModule_43301860278633
// MI455X (gfx1250) — matrix operand order repaired and checked by reading, not compiled, not run on hardware
//
// CDNA5 / gfx1250 implementation of SparseGraphLinkModule forward.
// All matmuls (~0.34 TFLOP total) go through v_wmma_f32_16x16x32_bf16 with
// f32 accumulation; bf16 operand staging uses global_load_async_to_lds_b128
// (ASYNCcnt) where it is a pure copy, and vectorized float4 -> v_cvt_pk_bf16
// -> ds_store_b128 where f32->bf16 conversion is needed. wave32, 256 thr/blk.

#include <hip/hip_runtime.h>
#include <math.h>

typedef __bf16 bf16;
typedef __attribute__((ext_vector_type(16))) __bf16 v16bf;
typedef __attribute__((ext_vector_type(8)))  __bf16 v8bf;
typedef __attribute__((ext_vector_type(8)))  float  v8f;

static constexpr int Bn  = 32;
static constexpr int NVn = 256;
static constexpr int NKn = 512;
static constexpr int Dn  = 1024;
static constexpr int Hn  = 8;
static constexpr int HDn = 128;
static constexpr float LN_EPS = 1e-5f;
static constexpr float ATT_SCALE = 0.08838834764831845f; // 1/sqrt(128)

union fragu { v16bf v; v8bf h[2]; };
union pk8u  { v8bf v; bf16 e[8]; };

// 32-byte global -> LDS async copy (CDNA5 VGLOBAL async ops, ASYNCcnt).
__device__ __forceinline__ void async_copy32(const void* g, const void* l) {
  asm volatile(
      "global_load_async_to_lds_b128 %0, %1, off\n\t"
      "global_load_async_to_lds_b128 %0, %1, off offset:16"
      :: "v"((unsigned)(size_t)l), "v"((unsigned long long)(size_t)g)
      : "memory");
}
__device__ __forceinline__ void async_wait0() {
  asm volatile("s_wait_asynccnt 0" ::: "memory");
}

// pack 8 floats (x zero-mask) into 8 bf16
__device__ __forceinline__ v8bf cvt8(float4 a, float4 b, float zm) {
  pk8u u;
  u.e[0] = (bf16)(a.x * zm); u.e[1] = (bf16)(a.y * zm);
  u.e[2] = (bf16)(a.z * zm); u.e[3] = (bf16)(a.w * zm);
  u.e[4] = (bf16)(b.x * zm); u.e[5] = (bf16)(b.y * zm);
  u.e[6] = (bf16)(b.z * zm); u.e[7] = (bf16)(b.w * zm);
  return u.v;
}

// ---------------------------------------------------------------------------
// Generic tiled WMMA GEMM:  C[z] = A[z] @ B[z] (+ bias) for z in grid.z
//   zo = z / zdiv, zi = z % zdiv;  base offsets = zo*s?o + zi*s?i (elements)
//   ABF : A is bf16 (pure async copy; caller guarantees M % 128 == 0)
//   BBT : B is [N][K] row-major (contraction contiguous); else B is [K][N]
//   BBF : B is bf16
//   CBF : C stored bf16;  GEL : exact GELU epilogue
// Tile 128x128x32, 256 threads = 8 waves (4 in M x 2 in N), each wave does a
// 32x64 sub-tile = 2x4 WMMA tiles, K-step 32. Needs K%32==0, N%128==0.
// ---------------------------------------------------------------------------
template<bool ABF, bool BBT, bool BBF, bool CBF, bool GEL>
__global__ __launch_bounds__(256) void gemm_k(
    const void* __restrict__ Ap, const void* __restrict__ Bp,
    const float* __restrict__ bias, void* __restrict__ Cp,
    int M, int N, int K, int lda, int ldb, int ldc,
    long sAo, long sAi, long sBo, long sBi, long sCo, long sCi, int zdiv)
{
  __shared__ __align__(16) bf16 As[128][40];
  __shared__ __align__(16) bf16 Bs[128][40];   // Bt layout: [n][k]

  const int z  = blockIdx.z;
  const int zo = z / zdiv, zi = z - zo * zdiv;
  const long aoff = (long)zo * sAo + (long)zi * sAi;
  const long boff = (long)zo * sBo + (long)zi * sBi;
  const long coff = (long)zo * sCo + (long)zi * sCi;

  const int mb = blockIdx.y * 128;
  const int nb = blockIdx.x * 128;
  const int t  = threadIdx.x;
  const int wave = t >> 5, lane = t & 31, hl = lane >> 4, lr = lane & 15;
  const int wm = (wave >> 1) * 32;   // wave row base within tile (0..96)
  const int wn = (wave & 1) * 64;    // wave col base within tile (0 or 64)

  // staging coordinates (hoisted)
  const int arow = t >> 1, akq = (t & 1) * 16;     // A: 2 thr/row, 16 k each
  const int gr   = mb + arow;
  const bool aok = gr < M;
  const int grc  = aok ? gr : 0;                   // clamped, branchless mask
  const float zm = aok ? 1.0f : 0.0f;
  const int bkk = t >> 3, bnn = (t & 7) * 16;      // B NN: 1 k-row, 16 n each

  v8f acc[2][4] = {};

  for (int k0 = 0; k0 < K; k0 += 32) {
    // ---- stage A tile -> As[row][k] ----
    if constexpr (ABF) {
      const bf16* s = (const bf16*)Ap + aoff + (long)gr * lda + k0 + akq;
      async_copy32(s, &As[arow][akq]);
    } else {
      const float4* s4 =
          (const float4*)((const float*)Ap + aoff + (long)grc * lda + k0 + akq);
      const float4 f0 = s4[0], f1 = s4[1], f2 = s4[2], f3 = s4[3];
      *(v8bf*)&As[arow][akq]     = cvt8(f0, f1, zm);
      *(v8bf*)&As[arow][akq + 8] = cvt8(f2, f3, zm);
    }
    // ---- stage B tile -> Bs[n][k] (contraction contiguous) ----
    if constexpr (BBT) {
      if constexpr (BBF) {
        const bf16* s = (const bf16*)Bp + boff + (long)(nb + arow) * ldb + k0 + akq;
        async_copy32(s, &Bs[arow][akq]);
      } else {
        const float4* s4 =
            (const float4*)((const float*)Bp + boff + (long)(nb + arow) * ldb + k0 + akq);
        const float4 f0 = s4[0], f1 = s4[1], f2 = s4[2], f3 = s4[3];
        *(v8bf*)&Bs[arow][akq]     = cvt8(f0, f1, 1.0f);
        *(v8bf*)&Bs[arow][akq + 8] = cvt8(f2, f3, 1.0f);
      }
    } else {
      if constexpr (BBF) {
        const v8bf* s8 = (const v8bf*)((const bf16*)Bp + boff + (long)(k0 + bkk) * ldb + nb + bnn);
        pk8u u0, u1; u0.v = s8[0]; u1.v = s8[1];
        #pragma unroll
        for (int j = 0; j < 8; ++j) { Bs[bnn + j][bkk] = u0.e[j]; Bs[bnn + 8 + j][bkk] = u1.e[j]; }
      } else {
        const float4* s4 =
            (const float4*)((const float*)Bp + boff + (long)(k0 + bkk) * ldb + nb + bnn);
        const float4 f0 = s4[0], f1 = s4[1], f2 = s4[2], f3 = s4[3];
        pk8u u0, u1; u0.v = cvt8(f0, f1, 1.0f); u1.v = cvt8(f2, f3, 1.0f);
        #pragma unroll
        for (int j = 0; j < 8; ++j) { Bs[bnn + j][bkk] = u0.e[j]; Bs[bnn + 8 + j][bkk] = u1.e[j]; }
      }
    }
    if constexpr (ABF || (BBT && BBF)) async_wait0();
    __syncthreads();

    // ---- fragments per ISA 7.12.2 layouts ----
    // A 16x32 bf16: lane<16 -> M=lr, K = {0..7,16..23}; lane>=16 -> K={8..15,24..31}
    v16bf afr[2];
    #pragma unroll
    for (int i = 0; i < 2; ++i) {
      const int row = wm + i * 16 + lr;
      const int kb = hl * 8;
      fragu u;
      u.h[0] = *(const v8bf*)&As[row][kb];
      u.h[1] = *(const v8bf*)&As[row][kb + 16];
      afr[i] = u.v;
    }
    // B 32x16 bf16: lane<16 -> N=lr, K = {0..7,16..23}; lane>=16 -> K={8..15,24..31}
    v16bf bfr[4];
    #pragma unroll
    for (int j = 0; j < 4; ++j) {
      const int n = wn + j * 16 + lr;
      const int kb = hl * 8;
      fragu u;
      u.h[0] = *(const v8bf*)&Bs[n][kb];
      u.h[1] = *(const v8bf*)&Bs[n][kb + 16];
      bfr[j] = u.v;
    }
    #pragma unroll
    for (int i = 0; i < 2; ++i)
      #pragma unroll
      for (int j = 0; j < 4; ++j)
        acc[i][j] = __builtin_amdgcn_wmma_f32_16x16x32_bf16(
            false, afr[i], false, bfr[j], (short)0, acc[i][j], false, false);
    __syncthreads();
  }

  // ---- epilogue: C layout: elem r -> M = base + hl*8 + r, N = base + lr ----
  #pragma unroll
  for (int j = 0; j < 4; ++j) {
    const int nCol = nb + wn + j * 16 + lr;
    const float bv = bias ? bias[nCol] : 0.0f;
    #pragma unroll
    for (int i = 0; i < 2; ++i) {
      #pragma unroll
      for (int r = 0; r < 8; ++r) {
        const int mRow = mb + wm + i * 16 + hl * 8 + r;
        if (mRow < M) {
          float v = acc[i][j][r] + bv;
          if constexpr (GEL) v = 0.5f * v * (1.0f + erff(v * 0.70710678118654752f));
          if constexpr (CBF)
            ((bf16*)Cp)[coff + (long)mRow * ldc + nCol] = (bf16)v;
          else
            ((float*)Cp)[coff + (long)mRow * ldc + nCol] = v;
        }
      }
    }
  }
}

// ---------------------------------------------------------------------------
// Block reductions (256 threads)
// ---------------------------------------------------------------------------
__device__ __forceinline__ float blk_sum(float v, float* red) {
  const int t = threadIdx.x;
  red[t] = v; __syncthreads();
  for (int s = 128; s > 0; s >>= 1) { if (t < s) red[t] += red[t + s]; __syncthreads(); }
  float r = red[0]; __syncthreads(); return r;
}
__device__ __forceinline__ float blk_max(float v, float* red) {
  const int t = threadIdx.x;
  red[t] = v; __syncthreads();
  for (int s = 128; s > 0; s >>= 1) { if (t < s) red[t] = fmaxf(red[t], red[t + s]); __syncthreads(); }
  float r = red[0]; __syncthreads(); return r;
}

// ---------------------------------------------------------------------------
// Fused (x + qbias[b]) -> LayerNorm -> L2-normalize -> bf16   (one row/block)
// ---------------------------------------------------------------------------
__global__ __launch_bounds__(256) void ln_l2_k(
    const float* __restrict__ X, const float* __restrict__ qb, int rowsPerBatch,
    const float* __restrict__ g, const float* __restrict__ bb, bf16* __restrict__ Y)
{
  __shared__ float red[256];
  const long row = blockIdx.x;
  const int b = (int)(row / rowsPerBatch);
  const int t = threadIdx.x;
  float x[4];
  #pragma unroll
  for (int i = 0; i < 4; ++i) {
    const int d = t + i * 256;
    x[i] = X[row * Dn + d] + qb[(long)b * Dn + d];
  }
  float s = x[0] + x[1] + x[2] + x[3];
  const float mu = blk_sum(s, red) * (1.0f / Dn);
  float vs = 0.f;
  #pragma unroll
  for (int i = 0; i < 4; ++i) { float d0 = x[i] - mu; vs += d0 * d0; }
  const float var = blk_sum(vs, red) * (1.0f / Dn);
  const float rstd = rsqrtf(var + LN_EPS);
  float y[4]; float n2 = 0.f;
  #pragma unroll
  for (int i = 0; i < 4; ++i) {
    const int d = t + i * 256;
    y[i] = (x[i] - mu) * rstd * g[d] + bb[d];
    n2 += y[i] * y[i];
  }
  const float nrm = sqrtf(blk_sum(n2, red));
  const float sc = 1.0f / fmaxf(nrm, 1e-12f);
  #pragma unroll
  for (int i = 0; i < 4; ++i) Y[row * Dn + t + i * 256] = (bf16)(y[i] * sc);
}

// ---------------------------------------------------------------------------
// out = mask ? LN(X + R) : R   (R/mask optional). Writes f32 row at ld/off.
// ---------------------------------------------------------------------------
__global__ __launch_bounds__(256) void res_ln_k(
    const float* __restrict__ X, const float* __restrict__ R,
    const unsigned char* __restrict__ msk,
    const float* __restrict__ g, const float* __restrict__ bb,
    float* __restrict__ out, int outLd, int outOff)
{
  __shared__ float red[256];
  const long row = blockIdx.x;
  const int t = threadIdx.x;
  float x[4];
  #pragma unroll
  for (int i = 0; i < 4; ++i) {
    const int d = t + i * 256;
    x[i] = X[row * Dn + d] + (R ? R[row * Dn + d] : 0.0f);
  }
  const float mu = blk_sum(x[0] + x[1] + x[2] + x[3], red) * (1.0f / Dn);
  float vs = 0.f;
  #pragma unroll
  for (int i = 0; i < 4; ++i) { float d0 = x[i] - mu; vs += d0 * d0; }
  const float rstd = rsqrtf(blk_sum(vs, red) * (1.0f / Dn) + LN_EPS);
  const bool keep = msk ? (msk[row] != 0) : true;
  #pragma unroll
  for (int i = 0; i < 4; ++i) {
    const int d = t + i * 256;
    const float yl = (x[i] - mu) * rstd * g[d] + bb[d];
    out[row * outLd + outOff + d] = keep ? yl : R[row * Dn + d];
  }
}

// ---------------------------------------------------------------------------
// Per-batch masked mean/std of scores -> low/high thresholds. (1 block/batch)
// ---------------------------------------------------------------------------
__global__ __launch_bounds__(256) void stats_k(
    const float* __restrict__ sc, const unsigned char* __restrict__ vm,
    const unsigned char* __restrict__ km, float* __restrict__ low, float* __restrict__ high)
{
  __shared__ float red[256];
  const int b = blockIdx.x;
  float s = 0.f, s2 = 0.f, c = 0.f;
  for (int i = threadIdx.x; i < NVn * NKn; i += 256) {
    const int vi = i / NKn, ki = i - vi * NKn;
    if (vm[b * NVn + vi] && km[b * NKn + ki]) {
      const float v = sc[(long)b * NVn * NKn + i];
      s += v; s2 += v * v; c += 1.f;
    }
  }
  s = blk_sum(s, red); s2 = blk_sum(s2, red); c = blk_sum(c, red);
  if (threadIdx.x == 0) {
    const float cnt = fmaxf(c, 1.f);
    const float mean = s / cnt;
    const float sd = sqrtf(fmaxf(s2 / cnt - mean * mean, 0.f));
    low[b]  = mean - 0.5f * sd;
    high[b] = mean + 0.5f * sd;
  }
}

// ---------------------------------------------------------------------------
// Per-row top-4 + rel weighting + softmax + renorm -> atomicMax scatter to cw
// ---------------------------------------------------------------------------
__global__ __launch_bounds__(256) void topk_k(
    const float* __restrict__ sc, const unsigned char* __restrict__ qm,
    const unsigned char* __restrict__ km, const float* __restrict__ low,
    const float* __restrict__ high, float* __restrict__ cw,
    int R, int C, long RS, long CS, long oRS, long oCS, int totalRows)
{
  const int row = blockIdx.x * 256 + threadIdx.x;
  if (row >= totalRows) return;
  const int b = row / R, r = row - b * R;
  const float* base = sc + (long)b * NVn * NKn + (long)r * RS;
  float tv[4] = {-INFINITY, -INFINITY, -INFINITY, -INFINITY};
  int   ti[4] = {-1, -1, -1, -1};
  for (int c = 0; c < C; ++c) {
    const float v = km[b * C + c] ? base[c * CS] : -INFINITY;
    if (v > tv[0]) { tv[3]=tv[2];ti[3]=ti[2]; tv[2]=tv[1];ti[2]=ti[1]; tv[1]=tv[0];ti[1]=ti[0]; tv[0]=v;ti[0]=c; }
    else if (v > tv[1]) { tv[3]=tv[2];ti[3]=ti[2]; tv[2]=tv[1];ti[2]=ti[1]; tv[1]=v;ti[1]=c; }
    else if (v > tv[2]) { tv[3]=tv[2];ti[3]=ti[2]; tv[2]=v;ti[2]=c; }
    else if (v > tv[3]) { tv[3]=v;ti[3]=c; }
  }
  const float lo = low[b], hi = high[b];
  float rel[4], w[4];
  float m2 = -INFINITY;
  #pragma unroll
  for (int i = 0; i < 4; ++i) {
    rel[i] = (tv[i] >= hi) ? 1.0f : ((tv[i] >= lo) ? 0.5f : 0.0f);
    if (rel[i] > 0.f) m2 = fmaxf(m2, tv[i]);
  }
  float Z = 0.f;
  #pragma unroll
  for (int i = 0; i < 4; ++i) { w[i] = (rel[i] > 0.f) ? expf(tv[i] - m2) : 0.f; Z += w[i]; }
  float ws = 0.f;
  #pragma unroll
  for (int i = 0; i < 4; ++i) { w[i] = (Z > 0.f) ? (w[i] / Z) * rel[i] : 0.f; ws += w[i]; }
  const float inv = 1.0f / fmaxf(ws, 1e-6f);
  if (!qm[b * R + r]) return;
  float* out = cw + (long)b * NVn * NKn + (long)r * oRS;
  #pragma unroll
  for (int i = 0; i < 4; ++i) {
    const float wi = w[i] * inv;
    if (wi > 0.f && ti[i] >= 0)
      atomicMax((int*)&out[(long)ti[i] * oCS], __float_as_int(wi));  // wi,cw >= 0
  }
}

// ---------------------------------------------------------------------------
// Weighted-softmax gating: P = exp(s*scale - rowmax)*ew (bf16), l = row sum.
// softmax->*ew->renorm collapses: final attn = P / l. One block per (b,h).
// ---------------------------------------------------------------------------
__global__ __launch_bounds__(256) void gate_k(
    const float* __restrict__ S, const float* __restrict__ cw,
    const unsigned char* __restrict__ qm, const unsigned char* __restrict__ km,
    bf16* __restrict__ P, float* __restrict__ lOut,
    int NQ, int NKk, long ewRS, long ewCS, float scale)
{
  const int z = blockIdx.x;
  const int b = z / Hn;
  for (int q = threadIdx.x; q < NQ; q += 256) {
    const float* sr = S + (long)z * NQ * NKk + (long)q * NKk;
    const float* ew = cw + (long)b * NVn * NKn + (long)q * ewRS;
    bf16* pr = P + (long)z * NQ * NKk + (long)q * NKk;
    float m = -INFINITY;
    if (qm[b * NQ + q]) {
      for (int k = 0; k < NKk; ++k)
        if (km[b * NKk + k] && ew[(long)k * ewCS] > 0.f)
          m = fmaxf(m, sr[k] * scale);
    }
    float l = 0.f;
    if (m > -INFINITY) {
      for (int k = 0; k < NKk; ++k) {
        const float e = km[b * NKk + k] ? ew[(long)k * ewCS] : 0.f;
        const float p = (e > 0.f) ? expf(sr[k] * scale - m) * e : 0.f;
        pr[k] = (bf16)p; l += p;
      }
    } else {
      for (int k = 0; k < NKk; ++k) pr[k] = (bf16)0.0f;
    }
    lOut[(long)z * NQ + q] = l;
  }
}

// o[b,q,h*HD+d] /= l[b,h,q]  (0 if l<=0)
__global__ __launch_bounds__(256) void divl_k(
    float* __restrict__ o, const float* __restrict__ l, int NQ, long total)
{
  const long i = (long)blockIdx.x * 256 + threadIdx.x;
  if (i >= total) return;
  const long b = i / ((long)NQ * Dn);
  const long rem = i - b * (long)NQ * Dn;
  const int q = (int)(rem / Dn);
  const int h = (int)((rem - (long)q * Dn) / HDn);
  const float lv = l[(b * Hn + h) * (long)NQ + q];
  o[i] = (lv > 0.f) ? o[i] / lv : 0.f;
}

// ---------------------------------------------------------------------------
// Masked softmax attention pooling: out[b, off..off+D) = sum_n w_n * nodes
// ---------------------------------------------------------------------------
__global__ __launch_bounds__(256) void pool_k(
    const float* __restrict__ nodes, const unsigned char* __restrict__ msk,
    const float* __restrict__ W, const float* __restrict__ b0, int N,
    float* __restrict__ out, int outLd, int outOff)
{
  __shared__ float scb[512];
  __shared__ float red[256];
  const int b = blockIdx.x;
  const float bias0 = b0[0];
  for (int n = threadIdx.x; n < N; n += 256) {
    float dot = 0.f;
    const float* row = nodes + ((long)b * N + n) * Dn;
    for (int d = 0; d < Dn; ++d) dot += row[d] * W[d];
    scb[n] = msk[b * N + n] ? (dot + bias0) : -INFINITY;
  }
  __syncthreads();
  float m = -INFINITY;
  for (int n = threadIdx.x; n < N; n += 256) m = fmaxf(m, scb[n]);
  m = blk_max(m, red);
  float s = 0.f;
  if (m > -INFINITY)
    for (int n = threadIdx.x; n < N; n += 256) s += expf(scb[n] - m);
  s = blk_sum(s, red);
  for (int n = threadIdx.x; n < N; n += 256)
    scb[n] = (m > -INFINITY && s > 0.f) ? expf(scb[n] - m) / s : 0.f;
  __syncthreads();
  for (int d = threadIdx.x; d < Dn; d += 256) {
    float acc = 0.f;
    for (int n = 0; n < N; ++n) acc += nodes[((long)b * N + n) * Dn + d] * scb[n];
    out[(long)b * outLd + outOff + d] = acc;
  }
}

// ===========================================================================
// Host side
// ===========================================================================
extern "C" void kernel_launch(void* const* d_in, const int* in_sizes, int n_in,
                              void* d_out, int out_size, void* d_ws, size_t ws_size,
                              hipStream_t stream) {
  (void)in_sizes; (void)n_in; (void)out_size; (void)ws_size;

  const float* visual = (const float*)d_in[0];
  const float* kg     = (const float*)d_in[1];
  const float* ques   = (const float*)d_in[2];
  const unsigned char* vmask = (const unsigned char*)d_in[3];
  const unsigned char* kmask = (const unsigned char*)d_in[4];
  const float *W_vis = (const float*)d_in[5],  *b_vis = (const float*)d_in[6];
  const float *W_kgs = (const float*)d_in[7],  *b_kgs = (const float*)d_in[8];
  const float *W_qs  = (const float*)d_in[9],  *b_qs  = (const float*)d_in[10];
  const float *W_sq  = (const float*)d_in[11], *b_sq  = (const float*)d_in[12];
  const float *W_sk  = (const float*)d_in[13], *b_sk  = (const float*)d_in[14];
  const float *W_sv  = (const float*)d_in[15], *b_sv  = (const float*)d_in[16];
  const float *W_kq  = (const float*)d_in[17], *b_kq  = (const float*)d_in[18];
  const float *W_kk  = (const float*)d_in[19], *b_kk  = (const float*)d_in[20];
  const float *W_kv  = (const float*)d_in[21], *b_kv  = (const float*)d_in[22];
  const float *W_so  = (const float*)d_in[23], *b_so  = (const float*)d_in[24];
  const float *W_ko  = (const float*)d_in[25], *b_ko  = (const float*)d_in[26];
  const float *W_l2  = (const float*)d_in[27], *b_l2  = (const float*)d_in[28];
  const float *W_sp  = (const float*)d_in[29], *b_sp  = (const float*)d_in[30];
  const float *W_kp  = (const float*)d_in[31], *b_kp  = (const float*)d_in[32];
  const float *W_l1  = (const float*)d_in[33], *b_l1  = (const float*)d_in[34];
  const float *g_sn  = (const float*)d_in[35], *be_sn = (const float*)d_in[36];
  const float *g_kn  = (const float*)d_in[37], *be_kn = (const float*)d_in[38];
  const float *g_qn  = (const float*)d_in[39], *be_qn = (const float*)d_in[40];

  // ---- workspace layout (needs ~390MB; persistent then phase-reused scratch)
  constexpr size_t MB = 1024ull * 1024ull;
  char* ws = (char*)d_ws;
  float* CW   = (float*)(ws + 0);            // B*NV*NK f32        16MB
  float* SCTX = (float*)(ws + 16 * MB);      // B*NV*D f32         32MB
  float* KCTX = (float*)(ws + 48 * MB);      // B*NK*D f32         64MB
  float* CAT  = (float*)(ws + 112 * MB);     // 32x3072 f32
  float* HBUF = (float*)(ws + 112 * MB + 512 * 1024);
  float* QB   = (float*)(ws + 112 * MB + 768 * 1024);  // q_bias 32x1024
  float* LOW  = (float*)(ws + 113 * MB);
  float* HIGH = (float*)(ws + 113 * MB + 256);
  float* LBUF = (float*)(ws + 113 * MB + 512);          // B*H*512 f32
  char*  SCR  = ws + 114 * MB;                          // phase scratch

  const dim3 blk(256);

  // ---------------- Phase 1: scoring path ----------------
  float* TMP    = (float*)(SCR + 0);        // up to 16384x1024 f32 (64MB)
  bf16*  VQ     = (bf16*)(SCR + 64 * MB);   // 8192x1024 bf16 (16MB)
  bf16*  KQ     = (bf16*)(SCR + 80 * MB);   // 16384x1024 bf16 (32MB)
  float* SCORES = (float*)(SCR + 112 * MB); // 32x256x512 f32 (16MB)

  gemm_k<false,false,false,false,false><<<dim3(8,1,1),blk,0,stream>>>(
      ques, W_qs, b_qs, QB, 32, Dn, Dn, Dn, Dn, Dn, 0,0,0,0,0,0, 1);
  gemm_k<false,false,false,false,false><<<dim3(8,64,1),blk,0,stream>>>(
      visual, W_vis, b_vis, TMP, Bn*NVn, Dn, Dn, Dn, Dn, Dn, 0,0,0,0,0,0, 1);
  ln_l2_k<<<Bn*NVn,blk,0,stream>>>(TMP, QB, NVn, g_sn, be_sn, VQ);
  gemm_k<false,false,false,false,false><<<dim3(8,128,1),blk,0,stream>>>(
      kg, W_kgs, b_kgs, TMP, Bn*NKn, Dn, Dn, Dn, Dn, Dn, 0,0,0,0,0,0, 1);
  ln_l2_k<<<Bn*NKn,blk,0,stream>>>(TMP, QB, NKn, g_kn, be_kn, KQ);
  // scores[b] = vq[b] @ kq[b]^T  (bf16 NT WMMA, async-to-LDS staging)
  gemm_k<true,true,true,false,false><<<dim3(4,2,Bn),blk,0,stream>>>(
      VQ, KQ, nullptr, SCORES, NVn, NKn, Dn, Dn, Dn, NKn,
      (long)NVn*Dn, 0, (long)NKn*Dn, 0, (long)NVn*NKn, 0, 1);
  stats_k<<<Bn,blk,0,stream>>>(SCORES, vmask, kmask, LOW, HIGH);
  hipMemsetAsync(CW, 0, (size_t)Bn*NVn*NKn*4, stream);
  topk_k<<<(Bn*NVn)/256,blk,0,stream>>>(SCORES, vmask, kmask, LOW, HIGH, CW,
      NVn, NKn, NKn, 1, NKn, 1, Bn*NVn);
  topk_k<<<(Bn*NKn)/256,blk,0,stream>>>(SCORES, kmask, vmask, LOW, HIGH, CW,
      NKn, NVn, 1, NKn, 1, NKn, Bn*NKn);

  // ---------------- Phase 2: scene cross-attention (visual <- kg) --------
  bf16*  QBF  = (bf16*)(SCR + 0);          // 8192x1024  (16MB)
  bf16*  KBF  = (bf16*)(SCR + 16 * MB);    // 16384x1024 (32MB)
  bf16*  VBF  = (bf16*)(SCR + 48 * MB);    // 16384x1024 (32MB)
  float* S1   = (float*)(SCR + 80 * MB);   // 32x8x256x512 f32 (128MB)
  bf16*  P1   = (bf16*)(SCR + 208 * MB);   // same, bf16 (64MB)
  float* O1   = (float*)(SCR + 80 * MB);   // reuse S1 region after gating
  float* PR1  = (float*)(SCR + 112 * MB);  // proj (32MB)

  gemm_k<false,false,false,true,false><<<dim3(8,64,1),blk,0,stream>>>(
      visual, W_sq, b_sq, QBF, Bn*NVn, Dn, Dn, Dn, Dn, Dn, 0,0,0,0,0,0, 1);
  gemm_k<false,false,false,true,false><<<dim3(8,128,1),blk,0,stream>>>(
      kg, W_sk, b_sk, KBF, Bn*NKn, Dn, Dn, Dn, Dn, Dn, 0,0,0,0,0,0, 1);
  gemm_k<false,false,false,true,false><<<dim3(8,128,1),blk,0,stream>>>(
      kg, W_sv, b_sv, VBF, Bn*NKn, Dn, Dn, Dn, Dn, Dn, 0,0,0,0,0,0, 1);
  gemm_k<true,true,true,false,false><<<dim3(4,2,Bn*Hn),blk,0,stream>>>(
      QBF, KBF, nullptr, S1, NVn, NKn, HDn, Dn, Dn, NKn,
      (long)NVn*Dn, HDn, (long)NKn*Dn, HDn, (long)Hn*NVn*NKn, (long)NVn*NKn, Hn);
  gate_k<<<Bn*Hn,blk,0,stream>>>(S1, CW, vmask, kmask, P1, LBUF,
      NVn, NKn, NKn, 1, ATT_SCALE);
  gemm_k<true,false,true,false,false><<<dim3(1,2,Bn*Hn),blk,0,stream>>>(
      P1, VBF, nullptr, O1, NVn, HDn, NKn, NKn, Dn, Dn,
      (long)Hn*NVn*NKn, (long)NVn*NKn, (long)NKn*Dn, HDn, (long)NVn*Dn, HDn, Hn);
  divl_k<<<(Bn*NVn*Dn)/256,blk,0,stream>>>(O1, LBUF, NVn, (long)Bn*NVn*Dn);
  gemm_k<false,false,false,false,false><<<dim3(8,64,1),blk,0,stream>>>(
      O1, W_so, b_so, PR1, Bn*NVn, Dn, Dn, Dn, Dn, Dn, 0,0,0,0,0,0, 1);
  res_ln_k<<<Bn*NVn,blk,0,stream>>>(PR1, visual, vmask, g_sn, be_sn, SCTX, Dn, 0);

  // ---------------- Phase 3: kg cross-attention (kg <- visual) -----------
  bf16*  QBF2 = (bf16*)(SCR + 0);          // 16384x1024 (32MB)
  bf16*  KBF2 = (bf16*)(SCR + 32 * MB);    // 8192x1024 (16MB)
  bf16*  VBF2 = (bf16*)(SCR + 48 * MB);    // 8192x1024 (16MB)
  float* S2   = (float*)(SCR + 64 * MB);   // 32x8x512x256 f32 (128MB)
  bf16*  P2   = (bf16*)(SCR + 192 * MB);   // 64MB
  float* O2   = (float*)(SCR + 64 * MB);   // reuse S2
  float* PR2  = (float*)(SCR + 128 * MB);  // 64MB

  gemm_k<false,false,false,true,false><<<dim3(8,128,1),blk,0,stream>>>(
      kg, W_kq, b_kq, QBF2, Bn*NKn, Dn, Dn, Dn, Dn, Dn, 0,0,0,0,0,0, 1);
  gemm_k<false,false,false,true,false><<<dim3(8,64,1),blk,0,stream>>>(
      visual, W_kk, b_kk, KBF2, Bn*NVn, Dn, Dn, Dn, Dn, Dn, 0,0,0,0,0,0, 1);
  gemm_k<false,false,false,true,false><<<dim3(8,64,1),blk,0,stream>>>(
      visual, W_kv, b_kv, VBF2, Bn*NVn, Dn, Dn, Dn, Dn, Dn, 0,0,0,0,0,0, 1);
  gemm_k<true,true,true,false,false><<<dim3(2,4,Bn*Hn),blk,0,stream>>>(
      QBF2, KBF2, nullptr, S2, NKn, NVn, HDn, Dn, Dn, NVn,
      (long)NKn*Dn, HDn, (long)NVn*Dn, HDn, (long)Hn*NKn*NVn, (long)NKn*NVn, Hn);
  gate_k<<<Bn*Hn,blk,0,stream>>>(S2, CW, kmask, vmask, P2, LBUF,
      NKn, NVn, 1, NKn, ATT_SCALE);   // ew transposed: cw[b, k(visual), q(kg)]
  gemm_k<true,false,true,false,false><<<dim3(1,4,Bn*Hn),blk,0,stream>>>(
      P2, VBF2, nullptr, O2, NKn, HDn, NVn, NVn, Dn, Dn,
      (long)Hn*NKn*NVn, (long)NKn*NVn, (long)NVn*Dn, HDn, (long)NKn*Dn, HDn, Hn);
  divl_k<<<(Bn*NKn*Dn)/256,blk,0,stream>>>(O2, LBUF, NKn, (long)Bn*NKn*Dn);
  gemm_k<false,false,false,false,false><<<dim3(8,128,1),blk,0,stream>>>(
      O2, W_ko, b_ko, PR2, Bn*NKn, Dn, Dn, Dn, Dn, Dn, 0,0,0,0,0,0, 1);
  res_ln_k<<<Bn*NKn,blk,0,stream>>>(PR2, kg, kmask, g_kn, be_kn, KCTX, Dn, 0);

  // ---------------- Phase 4: pooling, link MLP ----------------
  pool_k<<<Bn,blk,0,stream>>>(SCTX, vmask, W_sp, b_sp, NVn, CAT, 3*Dn, 0);
  pool_k<<<Bn,blk,0,stream>>>(KCTX, kmask, W_kp, b_kp, NKn, CAT, 3*Dn, Dn);
  res_ln_k<<<Bn,blk,0,stream>>>(ques, nullptr, nullptr, g_qn, be_qn, CAT, 3*Dn, 2*Dn);
  gemm_k<false,false,false,false,true><<<dim3(8,1,1),blk,0,stream>>>(
      CAT, W_l1, b_l1, HBUF, Bn, Dn, 3*Dn, 3*Dn, Dn, Dn, 0,0,0,0,0,0, 1);
  gemm_k<false,false,false,false,false><<<dim3(8,1,1),blk,0,stream>>>(
      HBUF, W_l2, b_l2, (float*)d_out, Bn, Dn, Dn, Dn, Dn, Dn, 0,0,0,0,0,0, 1);
}
